// RWKVTimeMix_54683523613239
// MI455X (gfx1250) — compile-verified, matrix operand order checked by reading, not run on hardware
//
#include <hip/hip_runtime.h>
#include <hip/hip_bf16.h>

typedef __attribute__((ext_vector_type(16))) __bf16 v16bf;
typedef __attribute__((ext_vector_type(8)))  __bf16 v8bf;
typedef __attribute__((ext_vector_type(4)))  __bf16 v4bf;
typedef __attribute__((ext_vector_type(8)))  float  v8f;

#define DIM   2048
#define TSEQ  2048
#define MROWS 8192      // B*T
#define BK    32
#define NTHREADS 256    // 8 wave32
#define NKSTEPS (DIM / BK)

// ---------------------------------------------------------------------------
// CDNA5 async global->LDS copy (ASYNCcnt path). The builtin's pointer params
// are `int __attribute__((vector_size(16)))*` in global/LDS address spaces
// (per hipcc's own diagnostic), so match that exactly.
// ---------------------------------------------------------------------------
#if defined(__has_builtin)
#if __has_builtin(__builtin_amdgcn_global_load_async_to_lds_b128)
#define HAVE_ASYNC_LDS 1
#endif
#endif
#ifndef HAVE_ASYNC_LDS
#define HAVE_ASYNC_LDS 0
#endif

#if HAVE_ASYNC_LDS
typedef int b128vec __attribute__((vector_size(16)));
#define AS_GLOBAL __attribute__((address_space(1)))
#define AS_LDS    __attribute__((address_space(3)))
__device__ __forceinline__ void copy16_async(const void* g, void* l) {
    __builtin_amdgcn_global_load_async_to_lds_b128(
        (AS_GLOBAL b128vec*)g, (AS_LDS b128vec*)l, 0, 0);
}
__device__ __forceinline__ void async_wait0() {
#if __has_builtin(__builtin_amdgcn_s_wait_asynccnt)
    __builtin_amdgcn_s_wait_asynccnt(0);
#else
    asm volatile("s_wait_asynccnt 0x0" ::: "memory");
#endif
}
#endif

__device__ __forceinline__ void copy16_sync(const void* g, void* l) {
    *(uint4*)l = *(const uint4*)g;
}

// ---------------------------------------------------------------------------
// Load a 16x32 bf16 WMMA operand fragment from a row-major LDS tile
// (row stride BK). Per ISA 7.12.2 (16-bit A/B layout):
//   lanes 0-15  : elems 0..7 -> K = 0..7,   elems 8..15 -> K = 16..23
//   lanes 16-31 : elems 0..7 -> K = 8..15,  elems 8..15 -> K = 24..31
// -> two contiguous ds_load_b128 per fragment.
// ---------------------------------------------------------------------------
__device__ __forceinline__ v16bf load_frag(const __bf16* base, int row, int half) {
    const __bf16* p = base + row * BK + half * 8;
    v8bf lo = *(const v8bf*)(p);
    v8bf hi = *(const v8bf*)(p + 16);
    return __builtin_shufflevector(lo, hi, 0,1,2,3,4,5,6,7,8,9,10,11,12,13,14,15);
}

__device__ __forceinline__ float fast_sigmoid(float x) {
    return __builtin_amdgcn_rcpf(1.0f + __expf(-x));
}

// ---------------------------------------------------------------------------
// Weight conversion f32 -> bf16 (4 weights, selected by blockIdx.z)
// ---------------------------------------------------------------------------
__global__ __launch_bounds__(256)
void wcvt_kernel(const float* __restrict__ w0, const float* __restrict__ w1,
                 const float* __restrict__ w2, const float* __restrict__ w3,
                 __bf16* __restrict__ o0, __bf16* __restrict__ o1,
                 __bf16* __restrict__ o2, __bf16* __restrict__ o3) {
    const float* w; __bf16* o;
    switch (blockIdx.z) {
        case 0:  w = w0; o = o0; break;
        case 1:  w = w1; o = o1; break;
        case 2:  w = w2; o = o2; break;
        default: w = w3; o = o3; break;
    }
    size_t i = ((size_t)blockIdx.x * blockDim.x + threadIdx.x) * 4;
    float4 f = *(const float4*)(w + i);
    v4bf b;
    b[0] = (__bf16)f.x; b[1] = (__bf16)f.y; b[2] = (__bf16)f.z; b[3] = (__bf16)f.w;
    *(v4bf*)(o + i) = b;
}

// ---------------------------------------------------------------------------
// Time-shift mix: xm = x * tc + x_prev * (1 - tc), stored bf16
// ---------------------------------------------------------------------------
__global__ __launch_bounds__(256)
void mix_kernel(const float* __restrict__ x, const float* __restrict__ tc,
                __bf16* __restrict__ xm) {
    size_t idx = ((size_t)blockIdx.x * blockDim.x + threadIdx.x) * 4;
    size_t m = idx / DIM;
    int    d = (int)(idx % DIM);
    int    t = (int)(m % TSEQ);
    float4 xv = *(const float4*)(x + idx);
    float4 xp;
    if (t == 0) { xp.x = 0.f; xp.y = 0.f; xp.z = 0.f; xp.w = 0.f; }
    else        { xp = *(const float4*)(x + idx - DIM); }
    float4 c = *(const float4*)(tc + d);
    v4bf o;
    o[0] = (__bf16)(xv.x * c.x + xp.x * (1.f - c.x));
    o[1] = (__bf16)(xv.y * c.y + xp.y * (1.f - c.y));
    o[2] = (__bf16)(xv.z * c.z + xp.z * (1.f - c.z));
    o[3] = (__bf16)(xv.w * c.w + xp.w * (1.f - c.w));
    *(v4bf*)(xm + idx) = o;
}

// ---------------------------------------------------------------------------
// Fused K/V/R GEMM + RWKV recurrence. Block tile 128x64, wave tile 32x32,
// 3 accumulator sets -> 12 wmma / K-step. Double-buffered LDS fed by
// async global->LDS copies: one barrier per K-step.
// ---------------------------------------------------------------------------
#define KVR_BM 128
#define KVR_BN 64

__global__ __launch_bounds__(NTHREADS)
void gemm_kvr_fused(const __bf16* __restrict__ xm,
                    const __bf16* __restrict__ wk,
                    const __bf16* __restrict__ wv,
                    const __bf16* __restrict__ wr,
                    const float*  __restrict__ state,
                    const float*  __restrict__ time_decay,
                    float*        __restrict__ new_state,
                    __bf16*       __restrict__ g) {
    __shared__ __align__(16) __bf16 As [2][KVR_BM * BK];
    __shared__ __align__(16) __bf16 Bks[2][KVR_BN * BK];
    __shared__ __align__(16) __bf16 Bvs[2][KVR_BN * BK];
    __shared__ __align__(16) __bf16 Brs[2][KVR_BN * BK];

    const int tid  = threadIdx.x;
    const int m0   = blockIdx.y * KVR_BM;
    const int n0   = blockIdx.x * KVR_BN;
    const int w    = tid >> 5;
    const int lane = tid & 31;
    const int half = lane >> 4;
    const int sub  = lane & 15;
    const int wm   = (w & 3) * 32;
    const int wn   = (w >> 2) * 32;

    // Per-thread copy-chunk geometry (16B chunks, row-major tiles, row=64B)
    const int cA0  = tid;              // A chunks 0..511 (2 per thread)
    const int cA1  = tid + NTHREADS;
    const int rA0  = cA0 >> 2, bA0 = (cA0 & 3) << 4;
    const int rA1  = cA1 >> 2, bA1 = (cA1 & 3) << 4;
    const int cB   = tid;              // B chunks 0..255 (1 per thread per weight)
    const int rB   = cB >> 2,  bB  = (cB & 3) << 4;

    v8f ak[2][2] = {}; v8f av[2][2] = {}; v8f ar[2][2] = {};

#define KVR_LOAD(kb, p, COPY)                                                        \
    do {                                                                             \
        const int k0 = (kb) * BK;                                                    \
        COPY((const char*)(xm + (size_t)(m0 + rA0) * DIM + k0) + bA0,                \
             (char*)As[p] + (size_t)cA0 * 16);                                       \
        COPY((const char*)(xm + (size_t)(m0 + rA1) * DIM + k0) + bA1,                \
             (char*)As[p] + (size_t)cA1 * 16);                                       \
        const size_t offB = (size_t)(n0 + rB) * DIM + k0;                            \
        COPY((const char*)(wk + offB) + bB, (char*)Bks[p] + (size_t)cB * 16);        \
        COPY((const char*)(wv + offB) + bB, (char*)Bvs[p] + (size_t)cB * 16);        \
        COPY((const char*)(wr + offB) + bB, (char*)Brs[p] + (size_t)cB * 16);        \
    } while (0)

#define KVR_COMPUTE(p)                                                               \
    do {                                                                             \
        v16bf af[2], bkf[2], bvf[2], brf[2];                                         \
        _Pragma("unroll")                                                            \
        for (int t = 0; t < 2; ++t) {                                                \
            af[t]  = load_frag(As[p],  wm + t * 16 + sub, half);                     \
            bkf[t] = load_frag(Bks[p], wn + t * 16 + sub, half);                     \
            bvf[t] = load_frag(Bvs[p], wn + t * 16 + sub, half);                     \
            brf[t] = load_frag(Brs[p], wn + t * 16 + sub, half);                     \
        }                                                                            \
        _Pragma("unroll")                                                            \
        for (int ti = 0; ti < 2; ++ti) {                                             \
            _Pragma("unroll")                                                        \
            for (int tj = 0; tj < 2; ++tj) {                                         \
                ak[ti][tj] = __builtin_amdgcn_wmma_f32_16x16x32_bf16(                \
                    false, af[ti], false, bkf[tj], (short)0, ak[ti][tj], false, false); \
                av[ti][tj] = __builtin_amdgcn_wmma_f32_16x16x32_bf16(                \
                    false, af[ti], false, bvf[tj], (short)0, av[ti][tj], false, false); \
                ar[ti][tj] = __builtin_amdgcn_wmma_f32_16x16x32_bf16(                \
                    false, af[ti], false, brf[tj], (short)0, ar[ti][tj], false, false); \
            }                                                                        \
        }                                                                            \
    } while (0)

#if HAVE_ASYNC_LDS
    int p = 0;
    KVR_LOAD(0, 0, copy16_async);
    async_wait0();
    __syncthreads();
    for (int kb = 0; kb < NKSTEPS; ++kb) {
        if (kb + 1 < NKSTEPS) KVR_LOAD(kb + 1, p ^ 1, copy16_async);
        KVR_COMPUTE(p);
        if (kb + 1 < NKSTEPS) {
            async_wait0();      // own async writes to buf p^1 landed
            __syncthreads();    // everyone's writes landed / reads of buf p done
            p ^= 1;
        }
    }
#else
    int p = 0;
    KVR_LOAD(0, 0, copy16_sync);
    __syncthreads();
    for (int kb = 0; kb < NKSTEPS; ++kb) {
        KVR_COMPUTE(p);
        if (kb + 1 < NKSTEPS) {
            __syncthreads();
            KVR_LOAD(kb + 1, p ^ 1, copy16_sync);
            __syncthreads();
            p ^= 1;
        }
    }
#endif

    // ---- fused RWKV epilogue ----
    #pragma unroll
    for (int tj = 0; tj < 2; ++tj) {
        const int e     = n0 + wn + tj * 16 + sub;
        const float dec = __expf(-__expf(time_decay[e]));
        #pragma unroll
        for (int ti = 0; ti < 2; ++ti) {
            #pragma unroll
            for (int i = 0; i < 8; ++i) {
                const int m = m0 + wm + ti * 16 + half * 8 + i;
                const size_t off = (size_t)m * DIM + e;
                const float ns = ak[ti][tj][i] * av[ti][tj][i] + dec * state[off];
                new_state[off] = ns;
                g[off] = (__bf16)(fast_sigmoid(ar[ti][tj][i]) * ns);
            }
        }
    }
#undef KVR_LOAD
#undef KVR_COMPUTE
}

// ---------------------------------------------------------------------------
// Output GEMM: out = g @ Wo^T. Block tile 128x128, wave tile 32x64
// (8 wmma per 12 ds_load_b128 per K-step), async double-buffered LDS.
// ---------------------------------------------------------------------------
#define OUT_BM 128
#define OUT_BN 128

__global__ __launch_bounds__(NTHREADS)
void gemm_out_kernel(const __bf16* __restrict__ g,
                     const __bf16* __restrict__ wo,
                     float* __restrict__ out) {
    __shared__ __align__(16) __bf16 As[2][OUT_BM * BK];
    __shared__ __align__(16) __bf16 Bs[2][OUT_BN * BK];

    const int tid  = threadIdx.x;
    const int m0   = blockIdx.y * OUT_BM;
    const int n0   = blockIdx.x * OUT_BN;
    const int w    = tid >> 5;
    const int lane = tid & 31;
    const int half = lane >> 4;
    const int sub  = lane & 15;
    const int wm   = (w & 3) * 32;    // 4 waves along M, 32 rows each
    const int wn   = (w >> 2) * 64;   // 2 waves along N, 64 cols each

    const int c0 = tid, c1 = tid + NTHREADS;   // 512 chunks each for A and B
    const int r0 = c0 >> 2, b0 = (c0 & 3) << 4;
    const int r1 = c1 >> 2, b1 = (c1 & 3) << 4;

    v8f acc[2][4] = {};

#define OUT_LOAD(kb, p, COPY)                                                        \
    do {                                                                             \
        const int k0 = (kb) * BK;                                                    \
        COPY((const char*)(g  + (size_t)(m0 + r0) * DIM + k0) + b0,                  \
             (char*)As[p] + (size_t)c0 * 16);                                        \
        COPY((const char*)(g  + (size_t)(m0 + r1) * DIM + k0) + b1,                  \
             (char*)As[p] + (size_t)c1 * 16);                                        \
        COPY((const char*)(wo + (size_t)(n0 + r0) * DIM + k0) + b0,                  \
             (char*)Bs[p] + (size_t)c0 * 16);                                        \
        COPY((const char*)(wo + (size_t)(n0 + r1) * DIM + k0) + b1,                  \
             (char*)Bs[p] + (size_t)c1 * 16);                                        \
    } while (0)

#define OUT_COMPUTE(p)                                                               \
    do {                                                                             \
        v16bf af[2], bf[4];                                                          \
        _Pragma("unroll")                                                            \
        for (int t = 0; t < 2; ++t)                                                  \
            af[t] = load_frag(As[p], wm + t * 16 + sub, half);                       \
        _Pragma("unroll")                                                            \
        for (int t = 0; t < 4; ++t)                                                  \
            bf[t] = load_frag(Bs[p], wn + t * 16 + sub, half);                       \
        _Pragma("unroll")                                                            \
        for (int ti = 0; ti < 2; ++ti) {                                             \
            _Pragma("unroll")                                                        \
            for (int tj = 0; tj < 4; ++tj)                                           \
                acc[ti][tj] = __builtin_amdgcn_wmma_f32_16x16x32_bf16(               \
                    false, af[ti], false, bf[tj], (short)0, acc[ti][tj], false, false); \
        }                                                                            \
    } while (0)

#if HAVE_ASYNC_LDS
    int p = 0;
    OUT_LOAD(0, 0, copy16_async);
    async_wait0();
    __syncthreads();
    for (int kb = 0; kb < NKSTEPS; ++kb) {
        if (kb + 1 < NKSTEPS) OUT_LOAD(kb + 1, p ^ 1, copy16_async);
        OUT_COMPUTE(p);
        if (kb + 1 < NKSTEPS) {
            async_wait0();
            __syncthreads();
            p ^= 1;
        }
    }
#else
    int p = 0;
    OUT_LOAD(0, 0, copy16_sync);
    __syncthreads();
    for (int kb = 0; kb < NKSTEPS; ++kb) {
        OUT_COMPUTE(p);
        if (kb + 1 < NKSTEPS) {
            __syncthreads();
            OUT_LOAD(kb + 1, p ^ 1, copy16_sync);
            __syncthreads();
            p ^= 1;
        }
    }
#endif

    #pragma unroll
    for (int ti = 0; ti < 2; ++ti) {
        #pragma unroll
        for (int tj = 0; tj < 4; ++tj) {
            const int e = n0 + wn + tj * 16 + sub;
            #pragma unroll
            for (int i = 0; i < 8; ++i) {
                const int m = m0 + wm + ti * 16 + half * 8 + i;
                out[(size_t)m * DIM + e] = acc[ti][tj][i];
            }
        }
    }
#undef OUT_LOAD
#undef OUT_COMPUTE
}

// ---------------------------------------------------------------------------
// Launch
// inputs: x, state, Wk, Wv, Wr, Wo, time_decay, time_curve (all f32)
// output: [out (8192*2048 f32) | new_state (8192*2048 f32)]
// ---------------------------------------------------------------------------
extern "C" void kernel_launch(void* const* d_in, const int* in_sizes, int n_in,
                              void* d_out, int out_size, void* d_ws, size_t ws_size,
                              hipStream_t stream) {
    const float* x     = (const float*)d_in[0];
    const float* state = (const float*)d_in[1];
    const float* Wk    = (const float*)d_in[2];
    const float* Wv    = (const float*)d_in[3];
    const float* Wr    = (const float*)d_in[4];
    const float* Wo    = (const float*)d_in[5];
    const float* td    = (const float*)d_in[6];
    const float* tc    = (const float*)d_in[7];

    char* ws = (char*)d_ws;
    const size_t WBYTES  = (size_t)DIM * DIM * 2;      // 8 MB per bf16 weight
    const size_t XMBYTES = (size_t)MROWS * DIM * 2;    // 32 MB
    __bf16* wk_bf = (__bf16*)(ws);
    __bf16* wv_bf = (__bf16*)(ws + WBYTES);
    __bf16* wr_bf = (__bf16*)(ws + 2 * WBYTES);
    __bf16* wo_bf = (__bf16*)(ws + 3 * WBYTES);
    __bf16* xm_bf = (__bf16*)(ws + 4 * WBYTES);
    __bf16* g_bf  = (__bf16*)(ws + 4 * WBYTES + XMBYTES);

    float* out_p = (float*)d_out;
    float* ns_p  = out_p + (size_t)MROWS * DIM;

    wcvt_kernel<<<dim3((DIM * DIM) / (256 * 4), 1, 4), 256, 0, stream>>>(
        Wk, Wv, Wr, Wo, wk_bf, wv_bf, wr_bf, wo_bf);

    mix_kernel<<<dim3(((size_t)MROWS * DIM) / (256 * 4)), 256, 0, stream>>>(x, tc, xm_bf);

    gemm_kvr_fused<<<dim3(DIM / KVR_BN, MROWS / KVR_BM), NTHREADS, 0, stream>>>(
        xm_bf, wk_bf, wv_bf, wr_bf, state, td, ns_p, g_bf);

    gemm_out_kernel<<<dim3(DIM / OUT_BN, MROWS / OUT_BM), NTHREADS, 0, stream>>>(
        g_bf, wo_bf, out_p);
}
